// MultiheadAttention_19026705121618
// MI455X (gfx1250) — matrix operand order repaired and checked by reading, not compiled, not run on hardware
//
#include <hip/hip_runtime.h>
#include <hip/hip_bf16.h>

// ---------------------------------------------------------------------------
// MHA on gfx1250 (MI455X): f16 WMMA (v_wmma_f32_16x16x32_f16), f32 accum.
// Pipeline: [f32->f16 staging] -> [qkv gemm] -> [causal flash attn] -> [out gemm]
// GEMM hot loops: batch all fragment loads into distinct registers, then issue
// 8 WMMAs -> staggered s_wait_loadcnt instead of full drains per WMMA pair.
// Working set (~64MB) stays L2-resident (192MB), so the cache path is the
// right data-movement strategy; no LDS staging of GEMM operands needed.
// ---------------------------------------------------------------------------

typedef __attribute__((ext_vector_type(16))) _Float16 v16h;
typedef __attribute__((ext_vector_type(8)))  _Float16 v8h;
typedef __attribute__((ext_vector_type(8)))  float    v8f;
typedef __attribute__((ext_vector_type(4)))  float    v4f;

#define D_MODEL 1024
#define HEADS   16
#define DK      64
#define BATCH   2
#define SEQ     2048

// ---- fragment helpers (layouts per CDNA5 ISA 7.12.2) ----------------------

// A fragment (16x32 f16): a[0..15] = K[hi*16..+15] (contiguous, matches B)
__device__ __forceinline__ v16h load_a_f16(const _Float16* p) {
  v8h lo = *(const v8h*)p;
  v8h hh = *(const v8h*)(p + 8);
  return __builtin_shufflevector(lo, hh, 0, 1, 2, 3, 4, 5, 6, 7,
                                 8, 9, 10, 11, 12, 13, 14, 15);
}

// B fragment (32x16 f16): 16 contiguous halves at col-row + hi*16 (32B aligned)
__device__ __forceinline__ v16h load_b_f16(const _Float16* p) {
  return *(const v16h*)p;
}

__device__ __forceinline__ v8f wmma_f16(v16h a, v16h b, v8f c) {
  return __builtin_amdgcn_wmma_f32_16x16x32_f16(false, a, false, b,
                                                (short)0, c, false, false);
}

// ---------------------------------------------------------------------------
// Kernel 0: one-time f32 -> f16 staging of all inputs/weights.
// grid = (2048, 7), block = 256; 8 elements per thread.
// ---------------------------------------------------------------------------
struct CvtArgs {
  const float* src[7];
  _Float16* dst[7];
  int n8[7];  // number of 8-element vectors per tensor
};

__global__ __launch_bounds__(256) void cvt_f16_kernel(CvtArgs a) {
  const int y = blockIdx.y;
  const int idx = blockIdx.x * 256 + threadIdx.x;
  if (idx >= a.n8[y]) return;
  const float* s = a.src[y] + (size_t)idx * 8;
  v4f f0 = *(const v4f*)s;
  v4f f1 = *(const v4f*)(s + 4);
  v8h h;
#pragma unroll
  for (int i = 0; i < 4; ++i) {
    h[i] = (_Float16)f0[i];
    h[4 + i] = (_Float16)f1[i];
  }
  *(v8h*)(a.dst[y] + (size_t)idx * 8) = h;
}

// ---------------------------------------------------------------------------
// Kernel 1: Q/K/V projection, y = x @ W^T (all f16 operands, f32 accum).
// grid = (16, 16, 3), block = 256 (8 waves); wave -> 32x64 output tile
// (2 A fragments share 4 B fragments -> 8 WMMAs per 32-wide k-step).
// Q,K stored [b,h,s,dk] f16; V stored transposed [b,h,dk,s] f16.
// ---------------------------------------------------------------------------
__global__ __launch_bounds__(256) void qkv_gemm_kernel(
    const _Float16* __restrict__ Xq, const _Float16* __restrict__ Xk,
    const _Float16* __restrict__ Xv,
    const _Float16* __restrict__ Wqh, const _Float16* __restrict__ Wkh,
    const _Float16* __restrict__ Wvh,
    _Float16* __restrict__ Qh, _Float16* __restrict__ Kh,
    _Float16* __restrict__ Vt) {
  const int z = blockIdx.z;
  const _Float16* X = (z == 0) ? Xq : (z == 1) ? Xk : Xv;
  const _Float16* W = (z == 0) ? Wqh : (z == 1) ? Wkh : Wvh;

  const int lane = threadIdx.x & 31;
  const int wave = threadIdx.x >> 5;
  const int lm = lane & 15;
  const int hi = lane >> 4;

  const int m0 = blockIdx.x * 256 + wave * 32;  // rows (b*S+s)
  const int n0 = blockIdx.y * 64;               // cols (h*64+dk)

  v8f acc[2][4] = {};
  const _Float16* ar0 = X + (size_t)(m0 + lm) * D_MODEL + hi * 16;
  const _Float16* ar1 = ar0 + (size_t)16 * D_MODEL;
  const _Float16* br = W + (size_t)(n0 + lm) * D_MODEL + hi * 16;

#pragma unroll 4
  for (int kk = 0; kk < D_MODEL; kk += 32) {
    // issue all 12 b128 loads first (distinct regs -> staggered waits)
    v16h a0 = load_a_f16(ar0 + kk);
    v16h a1 = load_a_f16(ar1 + kk);
    v16h b0 = load_b_f16(br + kk);
    v16h b1 = load_b_f16(br + (size_t)16 * D_MODEL + kk);
    v16h b2 = load_b_f16(br + (size_t)32 * D_MODEL + kk);
    v16h b3 = load_b_f16(br + (size_t)48 * D_MODEL + kk);
    acc[0][0] = wmma_f16(a0, b0, acc[0][0]);
    acc[1][0] = wmma_f16(a1, b0, acc[1][0]);
    acc[0][1] = wmma_f16(a0, b1, acc[0][1]);
    acc[1][1] = wmma_f16(a1, b1, acc[1][1]);
    acc[0][2] = wmma_f16(a0, b2, acc[0][2]);
    acc[1][2] = wmma_f16(a1, b2, acc[1][2]);
    acc[0][3] = wmma_f16(a0, b3, acc[0][3]);
    acc[1][3] = wmma_f16(a1, b3, acc[1][3]);
  }

#pragma unroll
  for (int g = 0; g < 2; ++g) {
#pragma unroll
    for (int r = 0; r < 8; ++r) {
      const int m = m0 + g * 16 + r + hi * 8;  // C layout: row = r + hi*8
      const int b = m >> 11;
      const int s = m & (SEQ - 1);
#pragma unroll
      for (int t = 0; t < 4; ++t) {
        const int n = n0 + t * 16 + lm;  // C layout: col = lane&15
        const int h = n >> 6;
        const int dk = n & (DK - 1);
        const _Float16 v = (_Float16)acc[g][t][r];
        if (z == 0)
          Qh[((size_t)(b * HEADS + h) * SEQ + s) * DK + dk] = v;
        else if (z == 1)
          Kh[((size_t)(b * HEADS + h) * SEQ + s) * DK + dk] = v;
        else
          Vt[((size_t)(b * HEADS + h) * DK + dk) * SEQ + s] = v;
      }
    }
  }
}

// ---------------------------------------------------------------------------
// Kernel 2: causal flash attention.
// grid = (16, 32), block = 256: wave -> one 16-query tile of one (b,h).
// Online softmax over 32-key blocks; P transposed C->A layout through LDS.
// ---------------------------------------------------------------------------
__global__ __launch_bounds__(256) void flash_attn_kernel(
    const _Float16* __restrict__ Qh, const _Float16* __restrict__ Kh,
    const _Float16* __restrict__ Vt, _Float16* __restrict__ AO) {
  __shared__ __align__(32) _Float16 pbuf[8][16][40];  // per-wave 16x32 P tile

  const int lane = threadIdx.x & 31;
  const int wave = threadIdx.x >> 5;
  const int lm = lane & 15;
  const int hi = lane >> 4;

  const int qt = blockIdx.x * 8 + wave;  // 0..127 query tile
  const int bh = blockIdx.y;             // 0..31
  const int qbase = qt * 16;

  const _Float16* Q = Qh + (size_t)bh * SEQ * DK;
  const _Float16* K = Kh + (size_t)bh * SEQ * DK;
  const _Float16* V = Vt + (size_t)bh * DK * SEQ;

  // persistent Q fragments: d = 0..31 and 32..63
  const _Float16* qp = Q + (size_t)(qbase + lm) * DK + hi * 16;
  const v16h qa0 = load_a_f16(qp);
  const v16h qa1 = load_a_f16(qp + 32);

  v8f o0 = {}, o1 = {}, o2 = {}, o3 = {};  // 16x64 O accumulator (f32)
  float mrow[8], lrow[8];
#pragma unroll
  for (int r = 0; r < 8; ++r) { mrow[r] = -1e30f; lrow[r] = 0.0f; }

  const float scale = 0.125f;              // 1/sqrt(64)
  const int nkb = (qbase + 16 + 31) >> 5;  // causal: only blocks <= diagonal

  for (int j = 0; j < nkb; ++j) {
    const int kb0 = j * 32;

    // logits S = Q @ K^T for 32 keys -> two 16x16 C tiles
    v8f st0 = {}, st1 = {};
    {
      const _Float16* kp0 = K + (size_t)(kb0 + lm) * DK + hi * 16;
      const _Float16* kp1 = K + (size_t)(kb0 + 16 + lm) * DK + hi * 16;
      v16h kb00 = load_b_f16(kp0);
      v16h kb01 = load_b_f16(kp0 + 32);
      v16h kb10 = load_b_f16(kp1);
      v16h kb11 = load_b_f16(kp1 + 32);
      st0 = wmma_f16(qa0, kb00, st0);
      st0 = wmma_f16(qa1, kb01, st0);
      st1 = wmma_f16(qa0, kb10, st1);
      st1 = wmma_f16(qa1, kb11, st1);
    }

    const bool need_mask = (j == nkb - 1);  // only diagonal block is partial
#pragma unroll
    for (int r = 0; r < 8; ++r) {
      const int row = r + hi * 8;
      const int query = qbase + row;
      float s0 = st0[r] * scale;
      float s1 = st1[r] * scale;
      if (need_mask) {
        if (kb0 + lm > query) s0 = -1e30f;
        if (kb0 + 16 + lm > query) s1 = -1e30f;
      }
      // row reductions across the 16-lane column group
      float rm = fmaxf(s0, s1);
#pragma unroll
      for (int d = 1; d < 16; d <<= 1) rm = fmaxf(rm, __shfl_xor(rm, d, 32));
      const float mnew = fmaxf(mrow[r], rm);
      const float alpha = __expf(mrow[r] - mnew);
      const float p0 = __expf(s0 - mnew);
      const float p1 = __expf(s1 - mnew);
      float ps = p0 + p1;
#pragma unroll
      for (int d = 1; d < 16; d <<= 1) ps += __shfl_xor(ps, d, 32);
      lrow[r] = lrow[r] * alpha + ps;
      mrow[r] = mnew;
      o0[r] *= alpha; o1[r] *= alpha; o2[r] *= alpha; o3[r] *= alpha;
      // C layout -> LDS, row-major 16x32 (padded to 40 halves)
      pbuf[wave][row][lm] = (_Float16)p0;
      pbuf[wave][row][lm + 16] = (_Float16)p1;
    }
    asm volatile("s_wait_dscnt 0" ::: "memory");  // LDS in-order per wave

    // reload P as an A fragment (16x32) and accumulate O += P @ V
    const v16h pa = load_a_f16(&pbuf[wave][lm][hi * 16]);
    const _Float16* vp = V + kb0 + hi * 16;  // Vt[dk][s]: contiguous in keys
    v16h vb0 = load_b_f16(vp + (size_t)(lm)*SEQ);
    v16h vb1 = load_b_f16(vp + (size_t)(16 + lm) * SEQ);
    v16h vb2 = load_b_f16(vp + (size_t)(32 + lm) * SEQ);
    v16h vb3 = load_b_f16(vp + (size_t)(48 + lm) * SEQ);
    o0 = wmma_f16(pa, vb0, o0);
    o1 = wmma_f16(pa, vb1, o1);
    o2 = wmma_f16(pa, vb2, o2);
    o3 = wmma_f16(pa, vb3, o3);
  }

  // normalize and store merged-heads layout [b, s, h*64+dk] (f16)
  const int b = bh >> 4;
  const int h = bh & 15;
#pragma unroll
  for (int r = 0; r < 8; ++r) {
    const int row = r + hi * 8;
    const float inv = 1.0f / lrow[r];
    _Float16* dst = AO + ((size_t)b * SEQ + qbase + row) * D_MODEL + h * DK;
    dst[lm]      = (_Float16)(o0[r] * inv);
    dst[16 + lm] = (_Float16)(o1[r] * inv);
    dst[32 + lm] = (_Float16)(o2[r] * inv);
    dst[48 + lm] = (_Float16)(o3[r] * inv);
  }
}

// ---------------------------------------------------------------------------
// Kernel 3: output projection  Y = AO @ Wo^T, f32 result.
// grid = (16, 16), block = 256; wave -> 32x64 tile (same shape as kernel 1).
// ---------------------------------------------------------------------------
__global__ __launch_bounds__(256) void out_proj_kernel(
    const _Float16* __restrict__ AO, const _Float16* __restrict__ Woh,
    float* __restrict__ Y) {
  const int lane = threadIdx.x & 31;
  const int wave = threadIdx.x >> 5;
  const int lm = lane & 15;
  const int hi = lane >> 4;
  const int m0 = blockIdx.x * 256 + wave * 32;
  const int n0 = blockIdx.y * 64;

  v8f acc[2][4] = {};
  const _Float16* ar0 = AO + (size_t)(m0 + lm) * D_MODEL + hi * 16;
  const _Float16* ar1 = ar0 + (size_t)16 * D_MODEL;
  const _Float16* br = Woh + (size_t)(n0 + lm) * D_MODEL + hi * 16;

#pragma unroll 4
  for (int kk = 0; kk < D_MODEL; kk += 32) {
    v16h a0 = load_a_f16(ar0 + kk);
    v16h a1 = load_a_f16(ar1 + kk);
    v16h b0 = load_b_f16(br + kk);
    v16h b1 = load_b_f16(br + (size_t)16 * D_MODEL + kk);
    v16h b2 = load_b_f16(br + (size_t)32 * D_MODEL + kk);
    v16h b3 = load_b_f16(br + (size_t)48 * D_MODEL + kk);
    acc[0][0] = wmma_f16(a0, b0, acc[0][0]);
    acc[1][0] = wmma_f16(a1, b0, acc[1][0]);
    acc[0][1] = wmma_f16(a0, b1, acc[0][1]);
    acc[1][1] = wmma_f16(a1, b1, acc[1][1]);
    acc[0][2] = wmma_f16(a0, b2, acc[0][2]);
    acc[1][2] = wmma_f16(a1, b2, acc[1][2]);
    acc[0][3] = wmma_f16(a0, b3, acc[0][3]);
    acc[1][3] = wmma_f16(a1, b3, acc[1][3]);
  }

#pragma unroll
  for (int g = 0; g < 2; ++g) {
#pragma unroll
    for (int r = 0; r < 8; ++r) {
      const int m = m0 + g * 16 + r + hi * 8;
#pragma unroll
      for (int t = 0; t < 4; ++t) {
        Y[(size_t)m * D_MODEL + n0 + t * 16 + lm] = acc[g][t][r];
      }
    }
  }
}

// ---------------------------------------------------------------------------
extern "C" void kernel_launch(void* const* d_in, const int* in_sizes, int n_in,
                              void* d_out, int out_size, void* d_ws,
                              size_t ws_size, hipStream_t stream) {
  (void)in_sizes; (void)n_in; (void)out_size; (void)ws_size;

  const float* q_src = (const float*)d_in[0];
  const float* k_src = (const float*)d_in[1];
  const float* v_src = (const float*)d_in[2];
  // d_in[3] = causal mask: implicit in the kernel (key <= query)
  const float* Wq = (const float*)d_in[4];
  const float* Wk = (const float*)d_in[5];
  const float* Wv = (const float*)d_in[6];
  const float* Wo = (const float*)d_in[7];

  const size_t XN = (size_t)BATCH * SEQ * D_MODEL;  // 4,194,304
  const size_t WN = (size_t)D_MODEL * D_MODEL;      // 1,048,576

  _Float16* p = (_Float16*)d_ws;
  _Float16* Xq = p;  p += XN;
  _Float16* Xk = p;  p += XN;
  _Float16* Xv = p;  p += XN;
  _Float16* Wqh = p; p += WN;
  _Float16* Wkh = p; p += WN;
  _Float16* Wvh = p; p += WN;
  _Float16* Woh = p; p += WN;
  _Float16* Qh = p;  p += XN;  // [b,h,s,dk]
  _Float16* Kh = p;  p += XN;  // [b,h,s,dk]
  _Float16* Vt = p;  p += XN;  // [b,h,dk,s]
  _Float16* AO = p;  p += XN;  // [b,s,h*dk]   total ws use = 64 MB

  CvtArgs ca;
  ca.src[0] = q_src; ca.dst[0] = Xq;  ca.n8[0] = (int)(XN / 8);
  ca.src[1] = k_src; ca.dst[1] = Xk;  ca.n8[1] = (int)(XN / 8);
  ca.src[2] = v_src; ca.dst[2] = Xv;  ca.n8[2] = (int)(XN / 8);
  ca.src[3] = Wq;    ca.dst[3] = Wqh; ca.n8[3] = (int)(WN / 8);
  ca.src[4] = Wk;    ca.dst[4] = Wkh; ca.n8[4] = (int)(WN / 8);
  ca.src[5] = Wv;    ca.dst[5] = Wvh; ca.n8[5] = (int)(WN / 8);
  ca.src[6] = Wo;    ca.dst[6] = Woh; ca.n8[6] = (int)(WN / 8);

  cvt_f16_kernel<<<dim3(2048, 7), 256, 0, stream>>>(ca);
  qkv_gemm_kernel<<<dim3(16, 16, 3), 256, 0, stream>>>(
      Xq, Xk, Xv, Wqh, Wkh, Wvh, Qh, Kh, Vt);
  flash_attn_kernel<<<dim3(16, 32), 256, 0, stream>>>(Qh, Kh, Vt, AO);
  out_proj_kernel<<<dim3(16, 16), 256, 0, stream>>>(AO, Woh, (float*)d_out);
}
